// ESIM_76390288327682
// MI455X (gfx1250) — matrix operand order repaired and checked by reading, not compiled, not run on hardware
//
#include <hip/hip_runtime.h>
#include <cstdint>
#include <cstddef>

// ---------------- types / WMMA helpers ----------------
typedef __attribute__((ext_vector_type(16))) __bf16 v16bf;
typedef __attribute__((ext_vector_type(8)))  __bf16 bf16x8;
typedef __attribute__((ext_vector_type(8)))  float  v8f;

#define WMMA_BF16(A, B, C) \
  __builtin_amdgcn_wmma_f32_16x16x32_bf16(false, (A), false, (B), (short)0, (C), false, false)

namespace esim {
constexpr int Ldim = 64;     // sequence length (LP == LH)
constexpr int Bdim = 128;    // batch
constexpr int Hdim = 512;    // hidden (== E)
constexpr int G4H  = 4 * Hdim;   // 2048 (gates)
constexpr int Ddim = 2 * Hdim;   // 1024 (bi-directional concat)
constexpr int Mdim = 4 * Ddim;   // 4096 (m_a / m_b feature width)
}

__device__ __forceinline__ float sigf(float x) { return 1.0f / (1.0f + expf(-x)); }

// A fragment: 16x32 bf16 tile from a row-major matrix (ld = row stride, elements).
// Lanes 0-15: row = lane,   K = k0..k0+15 contiguous
// Lanes 16-31: row = lane-16, K = k0+16..k0+31 contiguous
__device__ __forceinline__ v16bf load_fragA(const __bf16* base, long ld, int lane) {
  int r  = lane & 15;
  int kh = (lane & 16) ? 16 : 0;
  const __bf16* p = base + (long)r * ld + kh;
  union { v16bf v; bf16x8 h[2]; } u;
  u.h[0] = *(const bf16x8*)(p);
  u.h[1] = *(const bf16x8*)(p + 8);
  return u.v;
}

// B fragment for C = A @ W^T with W row-major (N x K).
// Lanes 0-15: col n = lane,    K = k0..k0+15 contiguous
// Lanes 16-31: col n = lane-16, K = k0+16..k0+31 contiguous
__device__ __forceinline__ v16bf load_fragB(const __bf16* base, long ld, int lane) {
  int n  = lane & 15;
  int kh = (lane & 16) ? 16 : 0;
  const __bf16* p = base + (long)n * ld + kh;
  union { v16bf v; bf16x8 h[2]; } u;
  u.h[0] = *(const bf16x8*)(p);
  u.h[1] = *(const bf16x8*)(p + 8);
  return u.v;
}

// B fragment where the K dimension is strided: element (n,k) at base[n + k*strideK]
__device__ __forceinline__ v16bf load_fragB_kstride(const __bf16* base, long strideK, int lane) {
  int n  = lane & 15;
  int kh = (lane & 16) ? 16 : 0;
  v16bf v;
#pragma unroll
  for (int i = 0; i < 16; ++i)
    v[i] = base[(long)(kh + i) * strideK + n];
  return v;
}

// ---------------- small utility kernels ----------------
__global__ void esim_cvt_bf16_kernel(const float* in, __bf16* out, long n) {
  long i = (long)blockIdx.x * blockDim.x + threadIdx.x;
  if (i < n) out[i] = (__bf16)in[i];
}

__global__ void esim_zero_kernel(float* p, long n) {
  long i = (long)blockIdx.x * blockDim.x + threadIdx.x;
  if (i < n) p[i] = 0.0f;
}

__global__ void esim_gather_embed_kernel(const int* tok, const float* embed, __bf16* x) {
  using namespace esim;
  int row = blockIdx.x;               // L*B rows
  int t = tok[row];
  const float* e = embed + (long)t * Hdim;   // E == Hdim
  __bf16* o = x + (long)row * Hdim;
  for (int k = threadIdx.x; k < Hdim; k += blockDim.x) o[k] = (__bf16)e[k];
}

// ---------------- fused LSTM step (GEMM over [x_t, h_prev] + cell update) ----------------
struct LstmDir {
  const __bf16* x;        // (L, B, I) bf16
  const __bf16* Wih;      // (4H, I)  bf16
  const __bf16* Whh;      // (4H, H)  bf16
  const float*  bias;     // (4H)     f32
  const __bf16* hprev;    // (B, H)   bf16
  __bf16*       hnew;     // (B, H)   bf16
  float*        cst;      // (B, H)   f32, updated in place
  float*        seq_f32;  // (L, B, 2H) output
  __bf16*       seq_bf16; // (L, B, 2H) output or nullptr
  int reverse;
  int dOff;               // 0 (fwd) or H (bwd) within the 2H concat
};
struct LstmStepParams { LstmDir dir[4]; int t, L, I; };

// ROWS = number of 16-row batch tiles per wave (1 for narrow stage, 2 for wide stage:
// reuses each weight fragment across 2 A tiles -> 1.5 b128 loads per WMMA instead of 2.5).
template <int ROWS>
__global__ __launch_bounds__(128) void esim_lstm_step_kernel(LstmStepParams P) {
  using namespace esim;
  LstmDir d = P.dir[blockIdx.z];
  int lane = threadIdx.x & 31;
  int tile = blockIdx.x * 4 + (threadIdx.x >> 5);   // (8/ROWS)*32 tiles per dir
  int rowBase = (tile >> 5) * (16 * ROWS);          // batch tile group
  int colBase = (tile & 31) * 16;                   // hidden tile (32 tiles)
  int tt = d.reverse ? (P.L - 1 - P.t) : P.t;

  v8f acc[ROWS][4] = {};

  // x_t @ Wih^T contribution
  const __bf16* xrow = d.x + (long)tt * Bdim * P.I + (long)rowBase * P.I;
  for (int k0 = 0; k0 < P.I; k0 += 32) {
    // pull the next A chunk toward the WGP while this k-step's WMMAs run
    __builtin_prefetch(xrow + k0 + 128, 0, 1);
    v16bf a[ROWS];
#pragma unroll
    for (int rr = 0; rr < ROWS; ++rr)
      a[rr] = load_fragA(xrow + (long)rr * 16 * P.I + k0, P.I, lane);
#pragma unroll
    for (int g = 0; g < 4; ++g) {
      v16bf w = load_fragB(d.Wih + (long)(g * Hdim + colBase) * P.I + k0, P.I, lane);
#pragma unroll
      for (int rr = 0; rr < ROWS; ++rr)
        acc[rr][g] = WMMA_BF16(a[rr], w, acc[rr][g]);
    }
  }
  // h_prev @ Whh^T contribution
  const __bf16* hrow = d.hprev + (long)rowBase * Hdim;
  for (int k0 = 0; k0 < Hdim; k0 += 32) {
    v16bf a[ROWS];
#pragma unroll
    for (int rr = 0; rr < ROWS; ++rr)
      a[rr] = load_fragA(hrow + (long)rr * 16 * Hdim + k0, Hdim, lane);
#pragma unroll
    for (int g = 0; g < 4; ++g) {
      v16bf w = load_fragB(d.Whh + (long)(g * Hdim + colBase) * Hdim + k0, Hdim, lane);
#pragma unroll
      for (int rr = 0; rr < ROWS; ++rr)
        acc[rr][g] = WMMA_BF16(a[rr], w, acc[rr][g]);
    }
  }

  // epilogue: gates -> cell update (each lane owns 8*ROWS (b,j) cells)
  int n = lane & 15;
  int j = colBase + n;
  int mOff = (lane & 16) ? 8 : 0;
  float bi = d.bias[j];
  float bfv = d.bias[Hdim + j];
  float bg = d.bias[2 * Hdim + j];
  float bo = d.bias[3 * Hdim + j];
#pragma unroll
  for (int rr = 0; rr < ROWS; ++rr) {
#pragma unroll
    for (int r = 0; r < 8; ++r) {
      int bb = rowBase + rr * 16 + mOff + r;
      float gi = sigf(acc[rr][0][r] + bi);
      float gf = sigf(acc[rr][1][r] + bfv);
      float gc = tanhf(acc[rr][2][r] + bg);
      float go = sigf(acc[rr][3][r] + bo);
      long ci = (long)bb * Hdim + j;
      float cn = gf * d.cst[ci] + gi * gc;
      float hv = go * tanhf(cn);
      d.cst[ci] = cn;
      d.hnew[ci] = (__bf16)hv;
      long so = (long)tt * Bdim * Ddim + (long)bb * Ddim + d.dOff + j;
      d.seq_f32[so] = hv;
      if (d.seq_bf16) d.seq_bf16[so] = (__bf16)hv;
    }
  }
}

// ---------------- attention ----------------
// scores[b,p,q] = sum_d prem[p,b,d] * hyp[q,b,d]   (d = 1024)
__global__ __launch_bounds__(128) void esim_scores_kernel(const __bf16* prem, const __bf16* hyp,
                                                          float* sc) {
  using namespace esim;
  int b = blockIdx.y;
  int lane = threadIdx.x & 31;
  int tile = blockIdx.x * 4 + (threadIdx.x >> 5);  // 0..15
  int pBase = (tile >> 2) * 16, qBase = (tile & 3) * 16;
  const long ldp = (long)Bdim * Ddim;              // stride of the sequence index
  const __bf16* arow = prem + (long)pBase * ldp + (long)b * Ddim;
  const __bf16* brow = hyp  + (long)qBase * ldp + (long)b * Ddim;
  v8f acc = {};
  for (int k0 = 0; k0 < Ddim; k0 += 32) {
    v16bf a = load_fragA(arow + k0, ldp, lane);
    v16bf w = load_fragB(brow + k0, ldp, lane);
    acc = WMMA_BF16(a, w, acc);
  }
  int n = lane & 15, mOff = (lane & 16) ? 8 : 0;
#pragma unroll
  for (int r = 0; r < 8; ++r)
    sc[(long)b * 4096 + (long)(pBase + mOff + r) * 64 + qBase + n] = acc[r];
}

// softmax over q for each (b,p); out[b,p,q] bf16
__global__ __launch_bounds__(128) void esim_softmax_rows_kernel(const float* sc, __bf16* out) {
  int lane = threadIdx.x & 31;
  int idx = blockIdx.x * 4 + (threadIdx.x >> 5);
  int b = idx >> 6, p = idx & 63;
  const float* row = sc + (long)b * 4096 + (long)p * 64;
  float v0 = row[lane], v1 = row[lane + 32];
  float m = fmaxf(v0, v1);
  for (int o = 16; o > 0; o >>= 1) m = fmaxf(m, __shfl_xor(m, o, 32));
  float e0 = expf(v0 - m), e1 = expf(v1 - m);
  float s = e0 + e1;
  for (int o = 16; o > 0; o >>= 1) s += __shfl_xor(s, o, 32);
  float inv = 1.0f / s;
  __bf16* orow = out + (long)b * 4096 + (long)p * 64;
  orow[lane] = (__bf16)(e0 * inv);
  orow[lane + 32] = (__bf16)(e1 * inv);
}

// softmax over p for each (b,q); written transposed: out[b,q,p] bf16
__global__ __launch_bounds__(128) void esim_softmax_cols_kernel(const float* sc, __bf16* outT) {
  int lane = threadIdx.x & 31;
  int idx = blockIdx.x * 4 + (threadIdx.x >> 5);
  int b = idx >> 6, q = idx & 63;
  const float* col = sc + (long)b * 4096 + q;
  float v0 = col[(long)lane * 64], v1 = col[(long)(lane + 32) * 64];
  float m = fmaxf(v0, v1);
  for (int o = 16; o > 0; o >>= 1) m = fmaxf(m, __shfl_xor(m, o, 32));
  float e0 = expf(v0 - m), e1 = expf(v1 - m);
  float s = e0 + e1;
  for (int o = 16; o > 0; o >>= 1) s += __shfl_xor(s, o, 32);
  float inv = 1.0f / s;
  __bf16* orow = outT + (long)b * 4096 + (long)q * 64;
  orow[lane] = (__bf16)(e0 * inv);
  orow[lane + 32] = (__bf16)(e1 * inv);
}

// out[m,b,n] = sum_k S[b,m,k] * V[k,b,n]   (k = 64, n = 1024)
__global__ __launch_bounds__(128) void esim_attn_apply_kernel(const __bf16* S, const __bf16* V,
                                                              float* out) {
  using namespace esim;
  int b = blockIdx.y;
  int lane = threadIdx.x & 31;
  int tile = blockIdx.x * 4 + (threadIdx.x >> 5);  // 0..255
  int mBase = (tile >> 6) * 16, nBase = (tile & 63) * 16;
  const long strideK = (long)Bdim * Ddim;
  const __bf16* srow = S + (long)b * 4096 + (long)mBase * 64;
  const __bf16* vbase = V + (long)b * Ddim + nBase;
  v8f acc = {};
  for (int k0 = 0; k0 < 64; k0 += 32) {
    v16bf a = load_fragA(srow + k0, 64, lane);
    v16bf w = load_fragB_kstride(vbase + (long)k0 * strideK, strideK, lane);
    acc = WMMA_BF16(a, w, acc);
  }
  int n = lane & 15, mOff = (lane & 16) ? 8 : 0;
#pragma unroll
  for (int r = 0; r < 8; ++r)
    out[(long)(mBase + mOff + r) * strideK + (long)b * Ddim + nBase + n] = acc[r];
}

// m[pb, 0:4096] = [seq, tilde, seq-tilde, seq*tilde] in bf16
__global__ void esim_concat_m_kernel(const float* seq, const float* tilde, __bf16* m, long n) {
  long i = (long)blockIdx.x * blockDim.x + threadIdx.x;
  if (i >= n) return;
  float a = seq[i], t = tilde[i];
  long pb = i >> 10;           // Ddim = 1024
  int dd = (int)(i & 1023);
  __bf16* row = m + pb * 4096;
  row[dd]        = (__bf16)a;
  row[1024 + dd] = (__bf16)t;
  row[2048 + dd] = (__bf16)(a - t);
  row[3072 + dd] = (__bf16)(a * t);
}

// v[b, 0:4096] = [mean(v1o), mean(v2o), max(v1o), max(v2o)] over time, bf16
__global__ void esim_pool_kernel(const float* v1o, const float* v2o, __bf16* v) {
  using namespace esim;
  int idx = blockIdx.x * blockDim.x + threadIdx.x;  // B * Ddim = 131072
  int b = idx >> 10, dd = idx & 1023;
  float s1 = 0.f, s2 = 0.f, m1 = -3.0e38f, m2 = -3.0e38f;
  for (int t = 0; t < Ldim; ++t) {
    long o = (long)t * Bdim * Ddim + (long)b * Ddim + dd;
    float a = v1o[o], c = v2o[o];
    s1 += a; s2 += c;
    m1 = fmaxf(m1, a); m2 = fmaxf(m2, c);
  }
  __bf16* row = v + (long)b * 4096;
  row[dd]        = (__bf16)(s1 * (1.0f / Ldim));
  row[1024 + dd] = (__bf16)(s2 * (1.0f / Ldim));
  row[2048 + dd] = (__bf16)m1;
  row[3072 + dd] = (__bf16)m2;
}

// hm = tanh(v @ mlp_W^T + mlp_b) : (128,4096) x (512,4096)^T
__global__ __launch_bounds__(128) void esim_mlp_kernel(const __bf16* v, const __bf16* W,
                                                       const float* bias, float* hm) {
  using namespace esim;
  int lane = threadIdx.x & 31;
  int tile = blockIdx.x * 4 + (threadIdx.x >> 5);  // 0..255
  int rBase = (tile >> 5) * 16, cBase = (tile & 31) * 16;
  v8f acc = {};
  for (int k0 = 0; k0 < Mdim; k0 += 32) {
    v16bf a = load_fragA(v + (long)rBase * Mdim + k0, Mdim, lane);
    v16bf w = load_fragB(W + (long)cBase * Mdim + k0, Mdim, lane);
    acc = WMMA_BF16(a, w, acc);
  }
  int n = lane & 15, mOff = (lane & 16) ? 8 : 0;
  int j = cBase + n;
  float bb = bias[j];
#pragma unroll
  for (int r = 0; r < 8; ++r)
    hm[(long)(rBase + mOff + r) * Hdim + j] = tanhf(acc[r] + bb);
}

// out = hm @ cl_W^T + cl_b : (128,512) x (3,512)^T -> (128,3)
__global__ __launch_bounds__(128) void esim_cls_kernel(const float* hm, const float* W,
                                                       const float* bias, float* out) {
  using namespace esim;
  int lane = threadIdx.x & 31;
  int idx = blockIdx.x * 4 + (threadIdx.x >> 5);
  if (idx >= Bdim * 3) return;
  int b = idx / 3, c = idx % 3;
  float s = 0.f;
  for (int k = lane; k < Hdim; k += 32) s += hm[(long)b * Hdim + k] * W[(long)c * Hdim + k];
  for (int o = 16; o > 0; o >>= 1) s += __shfl_xor(s, o, 32);
  if (lane == 0) out[b * 3 + c] = s + bias[c];
}

// ---------------- host orchestration ----------------
extern "C" void kernel_launch(void* const* d_in, const int* in_sizes, int n_in,
                              void* d_out, int out_size, void* d_ws, size_t ws_size,
                              hipStream_t stream) {
  using namespace esim;
  (void)in_sizes; (void)n_in; (void)out_size; (void)ws_size;

  char* ws = (char*)d_ws;
  size_t off = 0;
  auto alloc = [&](size_t bytes) -> void* {
    void* p = ws + off;
    off += (bytes + 255) & ~(size_t)255;
    return p;
  };

  const int*   tokP  = (const int*)d_in[0];
  const int*   tokH  = (const int*)d_in[1];
  const float* embed = (const float*)d_in[2];
  const float* mlp_W = (const float*)d_in[27];
  const float* mlp_b = (const float*)d_in[28];
  const float* cl_W  = (const float*)d_in[29];
  const float* cl_b  = (const float*)d_in[30];

  struct DirIn { int wih, whh, bias, I; };
  const DirIn din[8] = {
    {3, 4, 5, Hdim},  {6, 7, 8, Hdim},   {9, 10, 11, Hdim},  {12, 13, 14, Hdim},
    {15, 16, 17, Mdim},{18, 19, 20, Mdim},{21, 22, 23, Mdim}, {24, 25, 26, Mdim}};

  // ---- workspace layout ----
  __bf16* xp = (__bf16*)alloc((size_t)Ldim * Bdim * Hdim * 2);
  __bf16* xh = (__bf16*)alloc((size_t)Ldim * Bdim * Hdim * 2);
  __bf16* wih[8]; __bf16* whh[8];
  for (int i = 0; i < 8; ++i) {
    wih[i] = (__bf16*)alloc((size_t)G4H * din[i].I * 2);
    whh[i] = (__bf16*)alloc((size_t)G4H * Hdim * 2);
  }
  __bf16* mlpWb = (__bf16*)alloc((size_t)Hdim * Mdim * 2);

  __bf16* hbuf[8][2]; float* cbuf[8];
  size_t stateStart = off;
  for (int i = 0; i < 8; ++i) {
    hbuf[i][0] = (__bf16*)alloc((size_t)Bdim * Hdim * 2);
    hbuf[i][1] = (__bf16*)alloc((size_t)Bdim * Hdim * 2);
    cbuf[i]    = (float*) alloc((size_t)Bdim * Hdim * 4);
  }
  size_t stateBytes = off - stateStart;

  const size_t seqE = (size_t)Ldim * Bdim * Ddim;  // 8,388,608 elements
  float*  prem_f = (float*) alloc(seqE * 4);
  __bf16* prem_b = (__bf16*)alloc(seqE * 2);
  float*  hyp_f  = (float*) alloc(seqE * 4);
  __bf16* hyp_b  = (__bf16*)alloc(seqE * 2);
  float*  scores = (float*) alloc((size_t)Bdim * 64 * 64 * 4);
  __bf16* soft1  = (__bf16*)alloc((size_t)Bdim * 64 * 64 * 2);
  __bf16* soft0T = (__bf16*)alloc((size_t)Bdim * 64 * 64 * 2);
  float*  atilde = (float*) alloc(seqE * 4);
  float*  btilde = (float*) alloc(seqE * 4);
  __bf16* m_a    = (__bf16*)alloc((size_t)Ldim * Bdim * Mdim * 2);
  __bf16* m_b    = (__bf16*)alloc((size_t)Ldim * Bdim * Mdim * 2);
  float*  v1o    = (float*) alloc(seqE * 4);
  float*  v2o    = (float*) alloc(seqE * 4);
  __bf16* vpool  = (__bf16*)alloc((size_t)Bdim * Mdim * 2);
  float*  hm     = (float*) alloc((size_t)Bdim * Hdim * 4);

  // ---- stage A: conversions, gathers, state zeroing ----
  auto cvt = [&](const float* src, __bf16* dst, long n) {
    esim_cvt_bf16_kernel<<<dim3((unsigned)((n + 255) / 256)), dim3(256), 0, stream>>>(src, dst, n);
  };
  for (int i = 0; i < 8; ++i) {
    cvt((const float*)d_in[din[i].wih], wih[i], (long)G4H * din[i].I);
    cvt((const float*)d_in[din[i].whh], whh[i], (long)G4H * Hdim);
  }
  cvt(mlp_W, mlpWb, (long)Hdim * Mdim);

  esim_gather_embed_kernel<<<dim3(Ldim * Bdim), dim3(256), 0, stream>>>(tokP, embed, xp);
  esim_gather_embed_kernel<<<dim3(Ldim * Bdim), dim3(256), 0, stream>>>(tokH, embed, xh);

  long zn = (long)(stateBytes / 4);
  esim_zero_kernel<<<dim3((unsigned)((zn + 255) / 256)), dim3(256), 0, stream>>>(
      (float*)(ws + stateStart), zn);

  // ---- stage B: pe + he BiLSTMs (4 directions per launch, 64 sequential steps) ----
  // ROWS=1: small K (1024), keep 1024 waves in flight per launch for latency hiding.
  for (int t = 0; t < Ldim; ++t) {
    LstmStepParams P{};
    P.t = t; P.L = Ldim; P.I = Hdim;
    for (int j = 0; j < 4; ++j) {
      LstmDir& dd = P.dir[j];
      dd.x = (j < 2) ? xp : xh;
      dd.Wih = wih[j]; dd.Whh = whh[j];
      dd.bias = (const float*)d_in[din[j].bias];
      dd.hprev = hbuf[j][t & 1]; dd.hnew = hbuf[j][(t & 1) ^ 1];
      dd.cst = cbuf[j];
      dd.seq_f32  = (j < 2) ? prem_f : hyp_f;
      dd.seq_bf16 = (j < 2) ? prem_b : hyp_b;
      dd.reverse = j & 1; dd.dOff = (j & 1) * Hdim;
    }
    esim_lstm_step_kernel<1><<<dim3(64, 1, 4), dim3(128), 0, stream>>>(P);
  }

  // ---- stage C: attention ----
  esim_scores_kernel<<<dim3(4, Bdim), dim3(128), 0, stream>>>(prem_b, hyp_b, scores);
  esim_softmax_rows_kernel<<<dim3(Bdim * 64 / 4), dim3(128), 0, stream>>>(scores, soft1);
  esim_softmax_cols_kernel<<<dim3(Bdim * 64 / 4), dim3(128), 0, stream>>>(scores, soft0T);
  esim_attn_apply_kernel<<<dim3(64, Bdim), dim3(128), 0, stream>>>(soft1, hyp_b, atilde);
  esim_attn_apply_kernel<<<dim3(64, Bdim), dim3(128), 0, stream>>>(soft0T, prem_b, btilde);

  // ---- stage D: feature concat ----
  long cn = (long)seqE;
  esim_concat_m_kernel<<<dim3((unsigned)((cn + 255) / 256)), dim3(256), 0, stream>>>(
      prem_f, atilde, m_a, cn);
  esim_concat_m_kernel<<<dim3((unsigned)((cn + 255) / 256)), dim3(256), 0, stream>>>(
      hyp_f, btilde, m_b, cn);

  // ---- stage E: v1 + v2 BiLSTMs (I = 4096) ----
  // ROWS=2: compute-heavy stage; halve weight fragment traffic per WMMA.
  for (int t = 0; t < Ldim; ++t) {
    LstmStepParams P{};
    P.t = t; P.L = Ldim; P.I = Mdim;
    for (int j = 0; j < 4; ++j) {
      int di = 4 + j;
      LstmDir& dd = P.dir[j];
      dd.x = (j < 2) ? m_a : m_b;
      dd.Wih = wih[di]; dd.Whh = whh[di];
      dd.bias = (const float*)d_in[din[di].bias];
      dd.hprev = hbuf[di][t & 1]; dd.hnew = hbuf[di][(t & 1) ^ 1];
      dd.cst = cbuf[di];
      dd.seq_f32  = (j < 2) ? v1o : v2o;
      dd.seq_bf16 = nullptr;
      dd.reverse = j & 1; dd.dOff = (j & 1) * Hdim;
    }
    esim_lstm_step_kernel<2><<<dim3(32, 1, 4), dim3(128), 0, stream>>>(P);
  }

  // ---- stage F: pooling, MLP, classifier ----
  esim_pool_kernel<<<dim3(Bdim * Ddim / 256), dim3(256), 0, stream>>>(v1o, v2o, vpool);
  esim_mlp_kernel<<<dim3(64), dim3(128), 0, stream>>>(vpool, mlpWb, mlp_b, hm);
  esim_cls_kernel<<<dim3(96), dim3(128), 0, stream>>>(hm, cl_W, cl_b, (float*)d_out);
}
